// KOSLM_47528108098125
// MI455X (gfx1250) — matrix operand order repaired and checked by reading, not compiled, not run on hardware
//
#include <hip/hip_runtime.h>
#include <hip/hip_bf16.h>

#define Bc 4
#define Lc 128
#define Dm 512
#define Hc 192     // D_INNER
#define GH 384     // 2H
#define HB 576     // 3H

typedef __attribute__((ext_vector_type(16))) __bf16 v16bf;
typedef __attribute__((ext_vector_type(8)))  float  v8f;

union BF16Frag {
    v16bf v;
    unsigned short u[16];
};

__device__ __forceinline__ unsigned short f2bf(float f) {
    unsigned int u = __float_as_uint(f);
    u += 0x7FFFu + ((u >> 16) & 1u);           // round-to-nearest-even
    return (unsigned short)(u >> 16);
}

__device__ __forceinline__ float bflo(unsigned int p) { return __uint_as_float(p << 16); }
__device__ __forceinline__ float bfhi(unsigned int p) { return __uint_as_float(p & 0xFFFF0000u); }

__device__ __forceinline__ float sigf(float x) { return 1.0f / (1.0f + expf(-x)); }

__device__ __forceinline__ void load16(BF16Frag& f, const unsigned short* __restrict__ p) {
    const uint4* q = (const uint4*)p;          // 16B-aligned (k offsets are multiples of 16 bf16)
    ((uint4*)f.u)[0] = q[0];
    ((uint4*)f.u)[1] = q[1];
}

// load 8 consecutive floats (32B-aligned) via two b128 ops
__device__ __forceinline__ void ld8(const float* __restrict__ p, float* o) {
    float4 v0 = *(const float4*)p;
    float4 v1 = *(const float4*)(p + 4);
    o[0] = v0.x; o[1] = v0.y; o[2] = v0.z; o[3] = v0.w;
    o[4] = v1.x; o[5] = v1.y; o[6] = v1.z; o[7] = v1.w;
}

__device__ __forceinline__ v8f wmma_bf16(const BF16Frag& a, const BF16Frag& b, v8f c) {
    return __builtin_amdgcn_wmma_f32_16x16x32_bf16(false, a.v, false, b.v, (short)0, c,
                                                   false, false);
}

// ---------------- RMSNorm: one block per (b,l) ----------------
__global__ __launch_bounds__(256)
void rmsnorm_kernel(const float* __restrict__ x, const float* __restrict__ wnorm,
                    float* __restrict__ xn) {
    __shared__ float red[8];
    int bl = blockIdx.x, t = threadIdx.x;
    const float* xr = x + bl * Dm;
    float s = 0.f;
    for (int d = t; d < Dm; d += 256) { float v = xr[d]; s += v * v; }
    for (int off = 16; off > 0; off >>= 1) s += __shfl_down(s, off);
    if ((t & 31) == 0) red[t >> 5] = s;
    __syncthreads();
    if (t == 0) {
        float tot = 0.f;
        for (int i = 0; i < 8; ++i) tot += red[i];
        red[0] = rsqrtf(tot / (float)Dm + 1e-5f);
    }
    __syncthreads();
    float scale = red[0];
    for (int d = t; d < Dm; d += 256) xn[bl * Dm + d] = xr[d] * scale * wnorm[d];
}

// ------ causal depthwise conv (D_CONV=2) + SiLU + gates_x = xc@Wih^T + bias ------
__global__ __launch_bounds__(512)
void convgates_kernel(const float* __restrict__ xn, const float* __restrict__ cw,
                      const float* __restrict__ cb, const float* __restrict__ Wih,
                      const float* __restrict__ bih, const float* __restrict__ bhh,
                      float* __restrict__ gx) {
    __shared__ float xc[Dm];
    int bl = blockIdx.x;
    int ll = bl & (Lc - 1);
    int d = threadIdx.x;
    float prev = (ll == 0) ? 0.f : xn[(bl - 1) * Dm + d];
    float cur  = xn[bl * Dm + d];
    float v = cb[d] + cw[d * 2 + 0] * prev + cw[d * 2 + 1] * cur;
    xc[d] = v * sigf(v);                       // silu
    __syncthreads();
    if (d < GH) {
        float acc = bih[d] + bhh[d];
        const float4* wr = (const float4*)(Wih + d * Dm);
        const float4* xv = (const float4*)xc;
        for (int k = 0; k < Dm / 4; ++k) {
            float4 a = wr[k], b = xv[k];
            acc += a.x * b.x + a.y * b.y + a.z * b.z + a.w * b.w;
        }
        gx[bl * GH + d] = acc;
    }
}

// ---------------- r = xn @ in_w^T ----------------
__global__ __launch_bounds__(192)
void rproj_kernel(const float* __restrict__ xn, const float* __restrict__ inw,
                  float* __restrict__ r) {
    __shared__ float xs[Dm];
    int bl = blockIdx.x, t = threadIdx.x;
    for (int d = t; d < Dm; d += 192) xs[d] = xn[bl * Dm + d];
    __syncthreads();
    float acc = 0.f;
    const float4* wr = (const float4*)(inw + t * Dm);
    const float4* xv = (const float4*)xs;
    for (int d = 0; d < Dm / 4; ++d) {
        float4 a = wr[d], b = xv[d];
        acc += a.x * b.x + a.y * b.y + a.z * b.z + a.w * b.w;
    }
    r[bl * Hc + t] = acc;
}

// ---------------- fp32 -> bf16 weight conversion ----------------
__global__ void cvtw_kernel(const float* __restrict__ a, const float* __restrict__ b,
                            unsigned short* __restrict__ oa, unsigned short* __restrict__ ob,
                            int n) {
    int i = blockIdx.x * blockDim.x + threadIdx.x;
    if (i < n) { oa[i] = f2bf(a[i]); ob[i] = f2bf(b[i]); }
}

__global__ void cvt1_kernel(const float* __restrict__ a, unsigned short* __restrict__ o, int n) {
    int i = blockIdx.x * blockDim.x + threadIdx.x;
    if (i < n) o[i] = f2bf(a[i]);
}

// ---------------- sequential scan: one block per batch element ----------------
// 384 threads = 12 waves; wave w owns state rows 16w..16w+15 (wave-private rows).
// State kept as c and P = A*c (A itself is never needed: A_new = f(K)*Aconst and
// old A only appears via the product in innov = z - P*M). K = relu(innov@K1w^T+b1)
// @K2w^T+b2 runs on v_wmma_f32_16x16x32_bf16 with batched B-fragment loads; the
// h-reduction is done in registers with __shfl_xor instead of re-reading c from LDS.
__global__ __launch_bounds__(384)
void scan_kernel(const float* __restrict__ gx, const unsigned short* __restrict__ Whhbf,
                 const unsigned short* __restrict__ W1, const unsigned short* __restrict__ W2,
                 const float* __restrict__ k1b, const float* __restrict__ k2b,
                 const float* __restrict__ alog, const float* __restrict__ h0,
                 const float* __restrict__ c0, const float* __restrict__ A0,
                 float* __restrict__ y) {
    extern __shared__ char smem[];
    float* c_s    = (float*)smem;                // [H*H]
    float* P_s    = c_s + Hc * Hc;               // [H*H]  P = A*c
    float* z_s    = P_s + Hc * Hc;               // [H]
    float* m_s    = z_s + Hc;                    // [H]
    float* h_s    = m_s + Hc;                    // [H]   (16B aligned)
    float* ac_s   = h_s + Hc;                    // [H]   Acol = -exp(a_log[0,col])
    float* k1b_s  = ac_s + Hc;                   // [3H]
    float* k2b_s  = k1b_s + HB;                  // [H]
    float* hrow_s = k2b_s + Hc;                  // [H]   row sums of c_new
    unsigned short* stg = (unsigned short*)(hrow_s + Hc);  // 12 waves * 16*32 bf16

    const int tid = threadIdx.x, b = blockIdx.x;
    const int w = tid >> 5, l = tid & 31;
    const int r16 = l & 15, hi = l >> 4, seg = hi * 8;

    // ---- init state + constants ----
    for (int idx = tid; idx < Hc * Hc; idx += 384) {
        int m = idx / Hc;
        float cv = c0[b * Hc + m];               // c0 broadcast along last dim
        c_s[idx] = cv;
        P_s[idx] = A0[b * Hc * Hc + idx] * cv;
    }
    for (int idx = tid; idx < HB; idx += 384) k1b_s[idx] = k1b[idx];
    if (tid < Hc) {
        h_s[tid]   = h0[b * Hc + tid];
        ac_s[tid]  = -expf(alog[tid]);           // row 0 of a_log[layer] (all rows equal)
        k2b_s[tid] = k2b[tid];
    }
    __syncthreads();

    unsigned short* wstg = stg + w * 512;        // 16 rows x 32 k bf16 (row-major)
    const float* Prow = P_s + (16 * w + r16) * Hc;

    for (int t = 0; t < Lc; ++t) {
        // ---- phase 1: gates = gates_x + h @ Whh^T (bf16 weights) ; z/M ----
        {
            float acc = gx[(b * Lc + t) * GH + tid];
            if (t + 1 < Lc) __builtin_prefetch(&gx[(b * Lc + t + 1) * GH + tid], 0, 0);
            const uint4* wr = (const uint4*)(Whhbf + tid * Hc);   // 8 bf16 per uint4
            const float4* hv = (const float4*)h_s;
            for (int j = 0; j < Hc / 8; ++j) {
                uint4 wv = wr[j];
                float4 ha = hv[2 * j], hb = hv[2 * j + 1];
                acc += bflo(wv.x) * ha.x + bfhi(wv.x) * ha.y
                     + bflo(wv.y) * ha.z + bfhi(wv.y) * ha.w
                     + bflo(wv.z) * hb.x + bfhi(wv.z) * hb.y
                     + bflo(wv.w) * hb.z + bfhi(wv.w) * hb.w;
            }
            if (tid < Hc) z_s[tid] = tanhf(acc);
            else          m_s[tid - Hc] = sigf(acc);
        }
        __syncthreads();

        // ---- phase 2 (per-wave, rows 16w..16w+15) ----
        const int grow = 16 * w + r16;
        const float zr = z_s[grow], Mr = m_s[grow];

        // innov = z - P*M  as A-fragments for 6 k-chunks of 32 (bf16)
        BF16Frag innov[6];
        #pragma unroll
        for (int kc = 0; kc < 6; ++kc) {
            int kb = 32 * kc + hi * 16;
            float p0[8], p1[8];
            ld8(Prow + kb,      p0);
            ld8(Prow + kb + 8,  p1);
            #pragma unroll
            for (int e = 0; e < 8; ++e) {
                innov[kc].u[e]     = f2bf(zr - p0[e] * Mr);
                innov[kc].u[8 + e] = f2bf(zr - p1[e] * Mr);
            }
        }

        // K accumulators: 12 col-tiles, init with k2_b (column splat, from LDS)
        v8f Kacc[12];
        #pragma unroll
        for (int j = 0; j < 12; ++j) {
            float bv = k2b_s[16 * j + r16];
            v8f kv;
            #pragma unroll
            for (int e = 0; e < 8; ++e) kv[e] = bv;
            Kacc[j] = kv;
        }

        for (int kk = 0; kk < 18; ++kk) {        // GEMM2 k-chunks of 32 (= 2 GEMM1 n-tiles)
            // issue all 12 W2 B-fragment loads first: latency hidden behind GEMM1
            BF16Frag w2f[12];
            #pragma unroll
            for (int j = 0; j < 12; ++j)
                load16(w2f[j], W2 + (16 * j + r16) * HB + 32 * kk + hi * 16);

            #pragma unroll
            for (int sub = 0; sub < 2; ++sub) {
                const int n0 = 16 * (2 * kk + sub);
                BF16Frag w1f[6];
                #pragma unroll
                for (int kc = 0; kc < 6; ++kc)
                    load16(w1f[kc], W1 + (n0 + r16) * Hc + 32 * kc + hi * 16);
                float b1 = k1b_s[n0 + r16];
                v8f acc1;
                #pragma unroll
                for (int e = 0; e < 8; ++e) acc1[e] = b1;
                #pragma unroll
                for (int kc = 0; kc < 6; ++kc)
                    acc1 = wmma_bf16(innov[kc], w1f[kc], acc1);
                // relu -> bf16 -> staging tile (C-layout scatter)
                #pragma unroll
                for (int v = 0; v < 8; ++v) {
                    float xv = acc1[v] > 0.f ? acc1[v] : 0.f;
                    int m = v + 8 * hi;
                    wstg[m * 32 + sub * 16 + r16] = f2bf(xv);
                }
            }
            // reload staging as A-fragment (same-wave LDS ops are in order)
            BF16Frag tf;
            #pragma unroll
            for (int e = 0; e < 8; ++e) {
                tf.u[e]     = wstg[r16 * 32 + hi * 16 + e];
                tf.u[8 + e] = wstg[r16 * 32 + hi * 16 + 8 + e];
            }
            #pragma unroll
            for (int j = 0; j < 12; ++j)
                Kacc[j] = wmma_bf16(tf, w2f[j], Kacc[j]);
        }

        // ---- elementwise state update + in-register row sums ----
        float zrow[8], mrow[8], rsum[8];
        #pragma unroll
        for (int v = 0; v < 8; ++v) {
            int gm = 16 * w + v + 8 * hi;
            zrow[v] = z_s[gm];
            mrow[v] = m_s[gm];
            rsum[v] = 0.f;
        }
        #pragma unroll
        for (int j = 0; j < 12; ++j) {
            int col = 16 * j + r16;
            float Ac = ac_s[col];
            #pragma unroll
            for (int v = 0; v < 8; ++v) {
                int gm  = 16 * w + v + 8 * hi;
                int idx = gm * Hc + col;
                float Kv = Kacc[j][v];
                float An = sigf(((gm == col) ? 1.f : 0.f) - Kv * mrow[v]) * Ac;
                float cn = An * c_s[idx] + Kv * zrow[v];
                c_s[idx] = cn;
                P_s[idx] = An * cn;              // product state for next step's innov
                rsum[v] += cn;
            }
        }
        // reduce row sums across the 16 lanes sharing each row (xor stays in half)
        #pragma unroll
        for (int v = 0; v < 8; ++v) {
            float s = rsum[v];
            s += __shfl_xor(s, 1);
            s += __shfl_xor(s, 2);
            s += __shfl_xor(s, 4);
            s += __shfl_xor(s, 8);
            if (r16 == v) hrow_s[16 * w + v + 8 * hi] = s;
        }
        __syncthreads();

        // ---- phase 3: h_new = M * mean_j(c_new) ----
        if (tid < Hc) {
            float hn = m_s[tid] * hrow_s[tid] * (1.0f / (float)Hc);
            h_s[tid] = hn;
            y[(b * Lc + t) * Hc + tid] = hn;
        }
        __syncthreads();
    }
}

// ---------------- x_out = (y + r) @ out_w^T ----------------
__global__ __launch_bounds__(512)
void oproj_kernel(const float* __restrict__ y, const float* __restrict__ r,
                  const float* __restrict__ ow, float* __restrict__ xout) {
    __shared__ float s[Hc];
    int bl = blockIdx.x, t = threadIdx.x;
    if (t < Hc) s[t] = y[bl * Hc + t] + r[bl * Hc + t];
    __syncthreads();
    float acc = 0.f;
    const float4* wr = (const float4*)(ow + t * Hc);
    const float4* sv = (const float4*)s;
    for (int h = 0; h < Hc / 4; ++h) {
        float4 a = wr[h], b = sv[h];
        acc += a.x * b.x + a.y * b.y + a.z * b.z + a.w * b.w;
    }
    xout[bl * Dm + t] = acc;
}

extern "C" void kernel_launch(void* const* d_in, const int* in_sizes, int n_in,
                              void* d_out, int out_size, void* d_ws, size_t ws_size,
                              hipStream_t stream) {
    const float* x      = (const float*)d_in[0];
    const float* norm_w = (const float*)d_in[1];
    const float* conv_w = (const float*)d_in[2];
    const float* conv_b = (const float*)d_in[3];
    const float* in_w   = (const float*)d_in[4];
    const float* out_w  = (const float*)d_in[5];
    const float* w_ih   = (const float*)d_in[6];
    const float* w_hh   = (const float*)d_in[7];
    const float* b_ih   = (const float*)d_in[8];
    const float* b_hh   = (const float*)d_in[9];
    const float* a_log  = (const float*)d_in[10];
    const float* k1_w   = (const float*)d_in[11];
    const float* k1_b   = (const float*)d_in[12];
    const float* k2_w   = (const float*)d_in[13];
    const float* k2_b   = (const float*)d_in[14];
    const float* h0     = (const float*)d_in[15];
    const float* c0     = (const float*)d_in[16];
    const float* A0     = (const float*)d_in[17];
    float* out = (float*)d_out;

    // workspace carve (~4.3 MB)
    float* xn   = (float*)d_ws;
    float* gx   = xn   + Bc * Lc * Dm;
    float* rbuf = gx   + Bc * Lc * GH;
    float* ybuf = rbuf + Bc * Lc * Hc;
    float* xtmp = ybuf + Bc * Lc * Hc;
    unsigned short* K1bf  = (unsigned short*)(xtmp + Bc * Lc * Dm);
    unsigned short* K2bf  = K1bf + HB * Hc;
    unsigned short* Whhbf = K2bf + HB * Hc;

    const size_t scanLds = (size_t)(2 * Hc * Hc + 5 * Hc + HB) * sizeof(float)
                         + (size_t)(12 * 512) * sizeof(unsigned short);   // ~307 KB < 320 KB/WGP

    const float* xcur = x;
    for (int i = 0; i < 2; ++i) {
        rmsnorm_kernel<<<Bc * Lc, 256, 0, stream>>>(xcur, norm_w, xn);
        convgates_kernel<<<Bc * Lc, 512, 0, stream>>>(xn, conv_w + i * Dm * 2, conv_b + i * Dm,
                                                      w_ih + i * GH * Dm, b_ih + i * GH,
                                                      b_hh + i * GH, gx);
        rproj_kernel<<<Bc * Lc, 192, 0, stream>>>(xn, in_w + i * Hc * Dm, rbuf);
        cvtw_kernel<<<(HB * Hc + 255) / 256, 256, 0, stream>>>(k1_w + i * HB * Hc,
                                                               k2_w + i * Hc * HB,
                                                               K1bf, K2bf, HB * Hc);
        cvt1_kernel<<<(GH * Hc + 255) / 256, 256, 0, stream>>>(w_hh + i * GH * Hc, Whhbf,
                                                               GH * Hc);
        scan_kernel<<<Bc, 384, scanLds, stream>>>(gx, Whhbf, K1bf, K2bf,
                                                  k1_b + i * HB, k2_b + i * Hc,
                                                  a_log + i * Hc * Hc,
                                                  h0 + i * Bc * Hc, c0 + i * Bc * Hc,
                                                  A0 + i * Bc * Hc * Hc, ybuf);
        oproj_kernel<<<Bc * Lc, 512, 0, stream>>>(ybuf, rbuf, out_w + i * Dm * Hc,
                                                  (i == 0) ? xtmp : out);
        xcur = xtmp;
    }
}
